// NemotronHMoEW4A4Plugin_12360915878750
// MI455X (gfx1250) — matrix operand order repaired and checked by reading, not compiled, not run on hardware
//
#include <hip/hip_runtime.h>
#include <hip/hip_bf16.h>

// ---- model config (must match reference) ----
#define E_    16
#define TOPK_ 2
#define H_    2048
#define I_    1024
#define T_    4096          // B*S = 2*2048
#define GROUP_ 16

// GEMM blocking
#define KC_     64          // K chunk staged in LDS per iteration
#define MSUB_   4           // 16-row M subtiles per wave
#define WAVES_  8           // waves per workgroup (256 threads)
#define MBLK_   (WAVES_ * MSUB_ * 16)   // 512 rows per workgroup
#define MB_     (T_ / MBLK_)            // 8 M blocks covers worst case
#define UP_NT   (I_ / 16)   // 64
#define DN_NT   (H_ / 16)   // 128

typedef __attribute__((ext_vector_type(16))) _Float16 v16h;
typedef __attribute__((ext_vector_type(8)))  _Float16 v8h;
typedef __attribute__((ext_vector_type(8)))  float    v8f;
typedef __attribute__((ext_vector_type(4)))  unsigned int u32x4;
typedef __attribute__((ext_vector_type(8)))  unsigned int u32x8;

__device__ __forceinline__ v16h cat16(v8h lo, v8h hi) {
    return __builtin_shufflevector(lo, hi, 0,1,2,3,4,5,6,7,8,9,10,11,12,13,14,15);
}

// ---- Tensor Data Mover: async DMA of a 2D int32 tile (global -> LDS) ----
// D# per CDNA5 ISA (2-SGPR-group form, tensors <= 2D).
// All arguments must be workgroup-uniform (descriptor lives in SGPRs).
__device__ __forceinline__ void tdm_load_tile_2d(unsigned lds_off,            // LDS byte addr
                                                 unsigned long long gaddr,    // global byte addr
                                                 unsigned tensor_d0,          // row length (elems)
                                                 unsigned tensor_d1,          // #rows (elems)
                                                 unsigned long long stride0,  // row stride (elems)
                                                 unsigned tile_d0,            // tile cols (elems)
                                                 unsigned tile_d1)            // tile rows (elems)
{
    u32x4 g0;
    g0[0] = 1u;                                          // count=1, user descriptor
    g0[1] = lds_off;                                     // lds_addr (bytes)
    g0[2] = (unsigned)gaddr;                             // global_addr[31:0]
    g0[3] = (unsigned)(gaddr >> 32) | (2u << 30);        // global_addr[56:32] | type=2
    u32x8 g1;
    g1[0] = (2u << 16);                                  // data_size=4B; mask/flags=0
    g1[1] = (tensor_d0 & 0xFFFFu) << 16;                 // ab_addr=0 | tensor_dim0[15:0]
    g1[2] = (tensor_d0 >> 16) | ((tensor_d1 & 0xFFFFu) << 16);
    g1[3] = (tensor_d1 >> 16) | (tile_d0 << 16);         // tensor_dim1[31:16] | tile_dim0
    g1[4] = tile_d1;                                     // tile_dim1 | tile_dim2=0
    g1[5] = (unsigned)stride0;                           // tensor_dim0_stride[31:0]
    g1[6] = (unsigned)(stride0 >> 32);                   // stride0[47:32] | stride1[15:0]=0
    g1[7] = 0u;                                          // stride1 high
    asm volatile("tensor_load_to_lds %0, %1" :: "s"(g0), "s"(g1) : "memory");
}

__device__ __forceinline__ unsigned lds_offset_of(const void* p) {
    // generic LDS pointers carry the DS-usable offset in the low 32 bits
    return (unsigned)(unsigned long long)p;
}

// ---------------- kernel 0: zero output + counters ----------------
__global__ void moe_zero_kernel(float* __restrict__ y, int* __restrict__ counts, int n) {
    int idx = blockIdx.x * blockDim.x + threadIdx.x;
    if (idx < n) y[idx] = 0.0f;
    if (idx < E_) counts[idx] = 0;
}

// ---------------- kernel 0b: fp32 -> f16 activation pre-convert ----------------
__global__ void moe_cvt_kernel(const float* __restrict__ x, _Float16* __restrict__ xh) {
    int idx = blockIdx.x * blockDim.x + threadIdx.x;   // one float4 per thread
    float4 v = ((const float4*)x)[idx];
    _Float16* p = xh + (size_t)idx * 4;
    p[0] = (_Float16)v.x; p[1] = (_Float16)v.y;
    p[2] = (_Float16)v.z; p[3] = (_Float16)v.w;
}

// ---------------- kernel 1: router (one wave32 per token) ----------------
__global__ void moe_router_kernel(const float* __restrict__ x,
                                  const float* __restrict__ gate,   // [E,H]
                                  int* __restrict__ counts,         // [E]
                                  int* __restrict__ lists,          // [E,T]
                                  float* __restrict__ tokw)         // [E,T]
{
    int wave = blockIdx.x * (blockDim.x >> 5) + (threadIdx.x >> 5);
    int lane = threadIdx.x & 31;
    if (wave >= T_) return;
    int t = wave;

    const float* xp = x + (size_t)t * H_;
    float xr[H_ / 32];
#pragma unroll
    for (int j = 0; j < H_ / 32; ++j) xr[j] = xp[lane + 32 * j];

    float sig[E_];
#pragma unroll 1
    for (int e = 0; e < E_; ++e) {
        const float* gw = gate + (size_t)e * H_;
        float p = 0.0f;
#pragma unroll
        for (int j = 0; j < H_ / 32; ++j) p += xr[j] * gw[lane + 32 * j];
        for (int off = 16; off > 0; off >>= 1) p += __shfl_xor(p, off, 32);
        // softmax(logsigmoid(l)) == sigmoid(l)/sum(sigmoid(l)); top-2 renorm cancels sum
        sig[e] = 1.0f / (1.0f + __expf(-p));
    }

    int i0 = 0, i1 = -1;
    float v0 = -1.0f, v1 = -1.0f;
#pragma unroll
    for (int e = 0; e < E_; ++e) {
        float s = sig[e];
        if (s > v0) { v1 = v0; i1 = i0; v0 = s; i0 = e; }
        else if (s > v1) { v1 = s; i1 = e; }
    }
    float denom = v0 + v1 + 1e-20f;
    float w0 = v0 / denom;
    float w1 = v1 / denom;

    if (lane == 0) {
        int p0 = atomicAdd(&counts[i0], 1);
        lists[i0 * T_ + p0] = t;
        tokw [i0 * T_ + p0] = w0;
        int p1 = atomicAdd(&counts[i1], 1);
        lists[i1 * T_ + p1] = t;
        tokw [i1 * T_ + p1] = w1;
    }
}

// ---------------- kernel 2: exclusive prefix over expert counts ----------------
__global__ void moe_offsets_kernel(const int* __restrict__ counts, int* __restrict__ offsets) {
    if (threadIdx.x == 0 && blockIdx.x == 0) {
        int run = 0;
#pragma unroll
        for (int e = 0; e < E_; ++e) { offsets[e] = run; run += counts[e]; }
    }
}

// ---------------- kernel 3: up-projection, ReLU^2, routing-weight fold ----------------
// workgroup = (expert e, M block of 512 gathered rows, N tile of 16 columns).
// TDM double-buffers raw int4 code tiles [KC x 16] into LDS; 256 threads dequant
// codes -> f16 B tile in LDS; 8 waves x 4 M-subtiles x (KC/32) WMMAs per chunk.
__global__ void __launch_bounds__(256)
moe_up_kernel(const _Float16* __restrict__ xh,       // [T,H] f16
              const int*   __restrict__ qw,          // [E,H,I] int codes 0..15
              const float* __restrict__ bscale,      // [E,H/16,I]
              const float* __restrict__ gsp,         // [1]
              const int*   __restrict__ counts,
              const int*   __restrict__ offsets,
              const int*   __restrict__ lists,
              const float* __restrict__ tokw,
              _Float16*    __restrict__ act)         // [2T, I]
{
    __shared__ int      qs[2][KC_ * 16];             // raw code tiles [k][col], 2 x 4KB
    __shared__ _Float16 lb[16 * KC_];                // dequantized B: [col][k] f16, 2KB

    int b = blockIdx.x;
    int e  = b / (MB_ * UP_NT);
    int r0 = b - e * (MB_ * UP_NT);
    int mb = r0 / UP_NT;
    int n  = r0 - mb * UP_NT;

    int cnt = counts[e];
    if (mb * MBLK_ >= cnt) return;                   // uniform block early-exit

    int tid  = threadIdx.x;
    int wv   = tid >> 5;
    int lane = tid & 31;
    int half = lane >> 4;
    int l16  = lane & 15;

    // cooperative-dequant mapping: this thread owns (col dq_c, k's dq_k..dq_k+3)
    int dq_c = tid >> 4;              // 0..15
    int dq_k = (tid & 15) * 4;        // 0..60, stays inside one scale group of 16
    int colD = n * 16 + dq_c;
    const float* scolD = bscale + (size_t)e * (H_ / GROUP_) * I_ + colD;
    float gs = gsp[0];

    // TDM source geometry: tensor [H_ rows x I_ cols] int32, tile [KC_ x 16]
    unsigned long long wbase = (unsigned long long)(const void*)qw
                             + ((unsigned long long)e * H_ * I_ + (unsigned)(n * 16)) * 4ull;

    // wave's M rows
    int wavebase = mb * MBLK_ + wv * (MSUB_ * 16);
    bool active = wavebase < cnt;

    const _Float16* xrow[MSUB_];
#pragma unroll
    for (int t = 0; t < MSUB_; ++t) {
        int g = wavebase + t * 16 + l16;
        int gc = g < cnt ? g : cnt - 1;
        xrow[t] = xh + (size_t)lists[e * T_ + gc] * H_;
    }

    v8f c[MSUB_];
#pragma unroll
    for (int t = 0; t < MSUB_; ++t) c[t] = (v8f){};

    // prologue: DMA chunk 0 into buffer 0
    if (wv == 0)
        tdm_load_tile_2d(lds_offset_of(&qs[0][0]), wbase, I_, H_,
                         (unsigned long long)I_, 16u, KC_);

#pragma unroll 1
    for (int k0 = 0; k0 < H_; k0 += KC_) {
        int cur = (k0 / KC_) & 1;
        if (wv == 0) __builtin_amdgcn_s_wait_tensorcnt(0);   // chunk k0 landed
        __syncthreads();                                     // visible to all; lb free

        if (wv == 0 && k0 + KC_ < H_)                        // DMA next chunk
            tdm_load_tile_2d(lds_offset_of(&qs[cur ^ 1][0]),
                             wbase + (unsigned long long)(k0 + KC_) * I_ * 4ull,
                             I_, H_, (unsigned long long)I_, 16u, KC_);

        // ---- cooperative dequant: LDS codes -> f16 B tile
        {
            float sc = scolD[(size_t)((k0 + dq_k) >> 4) * I_] * gs;
            const int* qp = &qs[cur][dq_k * 16 + dq_c];
            int q0 = qp[0 * 16], q1 = qp[1 * 16], q2 = qp[2 * 16], q3 = qp[3 * 16];
            _Float16* lp = &lb[dq_c * KC_ + dq_k];
            lp[0] = (_Float16)((float)(q0 - 8) * sc);
            lp[1] = (_Float16)((float)(q1 - 8) * sc);
            lp[2] = (_Float16)((float)(q2 - 8) * sc);
            lp[3] = (_Float16)((float)(q3 - 8) * sc);
        }
        __syncthreads();

        if (active) {
#pragma unroll
            for (int ks = 0; ks < KC_; ks += 32) {
                // B (f16 32x16): lane holds column l16, K = ks+16*half .. +15
                const _Float16* lbp = &lb[l16 * KC_ + ks + 16 * half];
                v16h B = cat16(*(const v8h*)lbp, *(const v8h*)(lbp + 8));
#pragma unroll
                for (int t = 0; t < MSUB_; ++t) {
                    int ka = k0 + ks + 16 * half;
                    v16h A = cat16(*(const v8h*)(xrow[t] + ka),
                                   *(const v8h*)(xrow[t] + ka + 8));
                    c[t] = __builtin_amdgcn_wmma_f32_16x16x32_f16(
                               false, A, false, B, (short)0, c[t], false, false);
                }
            }
        }
    }

    // ---- epilogue: relu^2 * routing weight -> f16 gathered activation buffer
    if (active) {
        int offe = offsets[e];
#pragma unroll
        for (int t = 0; t < MSUB_; ++t) {
#pragma unroll
            for (int r = 0; r < 8; ++r) {
                int M = r + 8 * half;
                int g = wavebase + t * 16 + M;
                if (g < cnt) {
                    float w = tokw[e * T_ + g];
                    float z = c[t][r];
                    float rl = z > 0.0f ? z : 0.0f;
                    act[(size_t)(offe + g) * I_ + n * 16 + l16] = (_Float16)(rl * rl * w);
                }
            }
        }
    }
}

// ---------------- kernel 4: down-projection + scatter-add combine ----------------
__global__ void __launch_bounds__(256)
moe_down_kernel(const _Float16* __restrict__ act,    // [2T, I]
                const int*      __restrict__ qw,      // [E,I,H]
                const float*    __restrict__ bscale,  // [E,I/16,H]
                const float*    __restrict__ gsp,     // [1]
                const int*      __restrict__ counts,
                const int*      __restrict__ offsets,
                const int*      __restrict__ lists,
                float*          __restrict__ y)       // [T,H]
{
    __shared__ int      qs[2][KC_ * 16];
    __shared__ _Float16 lb[16 * KC_];

    int b = blockIdx.x;
    int e  = b / (MB_ * DN_NT);
    int r0 = b - e * (MB_ * DN_NT);
    int mb = r0 / DN_NT;
    int n  = r0 - mb * DN_NT;

    int cnt = counts[e];
    if (mb * MBLK_ >= cnt) return;

    int tid  = threadIdx.x;
    int wv   = tid >> 5;
    int lane = tid & 31;
    int half = lane >> 4;
    int l16  = lane & 15;

    int dq_c = tid >> 4;
    int dq_k = (tid & 15) * 4;
    int colD = n * 16 + dq_c;
    const float* scolD = bscale + (size_t)e * (I_ / GROUP_) * H_ + colD;
    float gs = gsp[0];

    unsigned long long wbase = (unsigned long long)(const void*)qw
                             + ((unsigned long long)e * I_ * H_ + (unsigned)(n * 16)) * 4ull;

    int offe = offsets[e];
    int wavebase = mb * MBLK_ + wv * (MSUB_ * 16);
    bool active = wavebase < cnt;

    const _Float16* arow[MSUB_];
#pragma unroll
    for (int t = 0; t < MSUB_; ++t) {
        int g = wavebase + t * 16 + l16;
        int gc = g < cnt ? g : cnt - 1;
        arow[t] = act + (size_t)(offe + gc) * I_;
    }

    v8f c[MSUB_];
#pragma unroll
    for (int t = 0; t < MSUB_; ++t) c[t] = (v8f){};

    if (wv == 0)
        tdm_load_tile_2d(lds_offset_of(&qs[0][0]), wbase, H_, I_,
                         (unsigned long long)H_, 16u, KC_);

#pragma unroll 1
    for (int k0 = 0; k0 < I_; k0 += KC_) {
        int cur = (k0 / KC_) & 1;
        if (wv == 0) __builtin_amdgcn_s_wait_tensorcnt(0);
        __syncthreads();

        if (wv == 0 && k0 + KC_ < I_)
            tdm_load_tile_2d(lds_offset_of(&qs[cur ^ 1][0]),
                             wbase + (unsigned long long)(k0 + KC_) * H_ * 4ull,
                             H_, I_, (unsigned long long)H_, 16u, KC_);

        {
            float sc = scolD[(size_t)((k0 + dq_k) >> 4) * H_] * gs;
            const int* qp = &qs[cur][dq_k * 16 + dq_c];
            int q0 = qp[0 * 16], q1 = qp[1 * 16], q2 = qp[2 * 16], q3 = qp[3 * 16];
            _Float16* lp = &lb[dq_c * KC_ + dq_k];
            lp[0] = (_Float16)((float)(q0 - 8) * sc);
            lp[1] = (_Float16)((float)(q1 - 8) * sc);
            lp[2] = (_Float16)((float)(q2 - 8) * sc);
            lp[3] = (_Float16)((float)(q3 - 8) * sc);
        }
        __syncthreads();

        if (active) {
#pragma unroll
            for (int ks = 0; ks < KC_; ks += 32) {
                const _Float16* lbp = &lb[l16 * KC_ + ks + 16 * half];
                v16h B = cat16(*(const v8h*)lbp, *(const v8h*)(lbp + 8));
#pragma unroll
                for (int t = 0; t < MSUB_; ++t) {
                    int ka = k0 + ks + 16 * half;
                    v16h A = cat16(*(const v8h*)(arow[t] + ka),
                                   *(const v8h*)(arow[t] + ka + 8));
                    c[t] = __builtin_amdgcn_wmma_f32_16x16x32_f16(
                               false, A, false, B, (short)0, c[t], false, false);
                }
            }
        }
    }

    if (active) {
#pragma unroll
        for (int t = 0; t < MSUB_; ++t) {
#pragma unroll
            for (int r = 0; r < 8; ++r) {
                int M = r + 8 * half;
                int g = wavebase + t * 16 + M;
                if (g < cnt) {
                    int tok = lists[e * T_ + g];
                    atomicAdd(&y[(size_t)tok * H_ + n * 16 + l16], c[t][r]);
                }
            }
        }
    }
}

// ---------------- host-side launcher ----------------
extern "C" void kernel_launch(void* const* d_in, const int* in_sizes, int n_in,
                              void* d_out, int out_size, void* d_ws, size_t ws_size,
                              hipStream_t stream) {
    (void)in_sizes; (void)n_in; (void)out_size; (void)ws_size;

    const float* x        = (const float*)d_in[0];  // [T,H]
    const float* gate     = (const float*)d_in[1];  // [E,H]
    const float* upscale  = (const float*)d_in[2];  // [E,H/16,I]
    const float* upgs     = (const float*)d_in[3];  // [1]
    const float* dnscale  = (const float*)d_in[4];  // [E,I/16,H]
    const float* dngs     = (const float*)d_in[5];  // [1]
    const int*   upq      = (const int*)d_in[6];    // [E,H,I]
    const int*   dnq      = (const int*)d_in[7];    // [E,I,H]
    float* y = (float*)d_out;                       // [T,H]

    // workspace layout
    char* ws = (char*)d_ws;
    int*      counts  = (int*)ws;                                   // 16 ints
    int*      offsets = (int*)(ws + 128);                           // 16 ints
    int*      lists   = (int*)(ws + 256);                           // E*T ints
    float*    tokw    = (float*)(ws + 256 + (size_t)E_ * T_ * 4);   // E*T floats
    _Float16* xh      = (_Float16*)(ws + 256 + (size_t)E_ * T_ * 8);        // [T,H] f16
    _Float16* act     = (_Float16*)(ws + 256 + (size_t)E_ * T_ * 8
                                        + (size_t)T_ * H_ * 2);             // [2T,I] f16

    // 0) zero output + expert counters
    {
        int n = T_ * H_;
        moe_zero_kernel<<<(n + 255) / 256, 256, 0, stream>>>(y, counts, n);
    }
    // 0b) fp32 -> f16 activations (vectorized, 4 per thread)
    moe_cvt_kernel<<<(T_ * H_ / 4 + 255) / 256, 256, 0, stream>>>(x, xh);
    // 1) router: 4096 waves, 8 waves / block
    moe_router_kernel<<<T_ / 8, 256, 0, stream>>>(x, gate, counts, lists, tokw);
    // 2) prefix
    moe_offsets_kernel<<<1, 32, 0, stream>>>(counts, offsets);
    // 3) up-proj: E * 8 * 64 workgroups
    moe_up_kernel<<<E_ * MB_ * UP_NT, 256, 0, stream>>>(xh, upq, upscale, upgs,
                                                        counts, offsets, lists, tokw, act);
    // 4) down-proj: E * 8 * 128 workgroups
    moe_down_kernel<<<E_ * MB_ * DN_NT, 256, 0, stream>>>(act, dnq, dnscale, dngs,
                                                          counts, offsets, lists, y);
}
